// SelfAttention_40913858461874
// MI455X (gfx1250) — matrix operand order repaired and checked by reading, not compiled, not run on hardware
//
#include <hip/hip_runtime.h>
#include <hip/hip_bf16.h>

// ---------------- problem constants ----------------
#define BATCH   2
#define L_SEQ   2048
#define CDIM    1024
#define H_HEADS 16
#define DHEAD   64
#define M1      (BATCH * L_SEQ)           // 4096 rows of x
#define RROWS   (BATCH * H_HEADS * L_SEQ) // 65536 head-rows per tensor

typedef _Float16 half_t;
typedef __attribute__((ext_vector_type(16))) _Float16 v16h;
typedef __attribute__((ext_vector_type(8)))  _Float16 v8h;
typedef __attribute__((ext_vector_type(8)))  float    v8f;
typedef __attribute__((ext_vector_type(4))) unsigned  u32x4;
typedef __attribute__((ext_vector_type(4))) int       i32x4;
typedef __attribute__((ext_vector_type(8))) int       i32x8;

// ---------------- WMMA fragment loaders (per CDNA5 ISA layouts) ----------------
// A-matrix 16x32 f16: lane m = lane&15, K: j<4 -> 8*hi+2j.., j>=4 -> 16+8*hi+2(j-4)..
// => two contiguous 16B chunks at (m*lda + 8*hi) and (+16 halves).
__device__ __forceinline__ v16h load_a_frag(const half_t* A, int lda) {
  const int lane = threadIdx.x & 31;
  const half_t* p = A + (size_t)(lane & 15) * lda + ((lane >> 4) << 3);
  v8h lo = *(const v8h*)(p);
  v8h hi = *(const v8h*)(p + 16);
  v16h f;
#pragma unroll
  for (int i = 0; i < 8; ++i) { f[i] = lo[i]; f[8 + i] = hi[i]; }
  return f;
}

// B-matrix 32x16 f16 from row-major [n][k] storage: lane n = lane&15,
// K: same as A => two contiguous 16B chunks at (n*ldb + 8*hi) and (+16 halves).
__device__ __forceinline__ v16h load_b_frag(const half_t* Bm, int ldb) {
  const int lane = threadIdx.x & 31;
  const half_t* p = Bm + (size_t)(lane & 15) * ldb + ((lane >> 4) << 3);
  v8h lo = *(const v8h*)(p);
  v8h hi = *(const v8h*)(p + 16);
  v16h f;
#pragma unroll
  for (int i = 0; i < 8; ++i) { f[i] = lo[i]; f[8 + i] = hi[i]; }
  return f;
}

__device__ __forceinline__ v8f wmma_f16(v16h a, v16h b, v8f c) {
  return __builtin_amdgcn_wmma_f32_16x16x32_f16(false, a, false, b, (short)0, c, false, false);
}

// ---------------- fp32 -> f16 conversion ----------------
__global__ __launch_bounds__(256) void cvt_f32_f16_kernel(const float* __restrict__ in,
                                                          half_t* __restrict__ out, int n) {
  int i = blockIdx.x * 256 + threadIdx.x;
  if (i < n) out[i] = (half_t)in[i];
}

// ---------------- shared 32x64-tile GEMM core, copy-free double buffering ----------------
// acc[2][4] covers rows m0..m0+31 x cols n0..n0+63; Wcol = W + n0*CDIM.
__device__ __forceinline__ void gemm_core_32x64(const half_t* __restrict__ Ab,
                                                const half_t* __restrict__ Wcol,
                                                v8f (&acc)[2][4]) {
  v16h a0[2], b0[4], a1[2], b1[4];
  a0[0] = load_a_frag(Ab, CDIM);
  a0[1] = load_a_frag(Ab + (size_t)16 * CDIM, CDIM);
#pragma unroll
  for (int j = 0; j < 4; ++j) b0[j] = load_b_frag(Wcol + (size_t)16 * j * CDIM, CDIM);

  for (int k = 0; k + 64 < CDIM; k += 64) {
    a1[0] = load_a_frag(Ab + k + 32, CDIM);
    a1[1] = load_a_frag(Ab + (size_t)16 * CDIM + k + 32, CDIM);
#pragma unroll
    for (int j = 0; j < 4; ++j) b1[j] = load_b_frag(Wcol + (size_t)16 * j * CDIM + k + 32, CDIM);
#pragma unroll
    for (int j = 0; j < 4; ++j) {
      acc[0][j] = wmma_f16(a0[0], b0[j], acc[0][j]);
      acc[1][j] = wmma_f16(a0[1], b0[j], acc[1][j]);
    }
    a0[0] = load_a_frag(Ab + k + 64, CDIM);
    a0[1] = load_a_frag(Ab + (size_t)16 * CDIM + k + 64, CDIM);
#pragma unroll
    for (int j = 0; j < 4; ++j) b0[j] = load_b_frag(Wcol + (size_t)16 * j * CDIM + k + 64, CDIM);
#pragma unroll
    for (int j = 0; j < 4; ++j) {
      acc[0][j] = wmma_f16(a1[0], b1[j], acc[0][j]);
      acc[1][j] = wmma_f16(a1[1], b1[j], acc[1][j]);
    }
  }
  // tail: buf0 holds k = CDIM-64; load buf1 @ CDIM-32
  a1[0] = load_a_frag(Ab + CDIM - 32, CDIM);
  a1[1] = load_a_frag(Ab + (size_t)16 * CDIM + CDIM - 32, CDIM);
#pragma unroll
  for (int j = 0; j < 4; ++j) b1[j] = load_b_frag(Wcol + (size_t)16 * j * CDIM + CDIM - 32, CDIM);
#pragma unroll
  for (int j = 0; j < 4; ++j) {
    acc[0][j] = wmma_f16(a0[0], b0[j], acc[0][j]);
    acc[1][j] = wmma_f16(a0[1], b0[j], acc[1][j]);
  }
#pragma unroll
  for (int j = 0; j < 4; ++j) {
    acc[0][j] = wmma_f16(a1[0], b1[j], acc[0][j]);
    acc[1][j] = wmma_f16(a1[1], b1[j], acc[1][j]);
  }
}

// ---------------- QKV GEMM: qkv = x @ W_qkv^T + bias, scattered to [t][b][h][l][d] ----------------
__global__ __launch_bounds__(32)
void qkv_gemm_kernel(const half_t* __restrict__ xh, const half_t* __restrict__ wh,
                     const float* __restrict__ qb, const float* __restrict__ vb,
                     half_t* __restrict__ qkvh) {
  const int m0 = blockIdx.x * 32;
  const int n0 = blockIdx.y * 64;
  const int lane = threadIdx.x & 31;
  const int ln = lane & 15, hif = lane >> 4;

  v8f zero = {};
  v8f acc[2][4];
#pragma unroll
  for (int mi = 0; mi < 2; ++mi)
#pragma unroll
    for (int j = 0; j < 4; ++j) acc[mi][j] = zero;

  gemm_core_32x64(xh + (size_t)m0 * CDIM, wh + (size_t)n0 * CDIM, acc);

#pragma unroll
  for (int j = 0; j < 4; ++j) {
    const int n = n0 + 16 * j + ln;
    const int t = n >> 10;           // which of q/k/v
    const int cc = n & 1023;
    const int hh = cc >> 6, dd = cc & 63;
    const float bias = (t == 0) ? qb[cc] : ((t == 2) ? vb[cc] : 0.0f);
#pragma unroll
    for (int mi = 0; mi < 2; ++mi)
#pragma unroll
      for (int r = 0; r < 8; ++r) {
        const int m = m0 + mi * 16 + r + 8 * hif;
        const int bb = m >> 11, l = m & 2047;
        const size_t row = (size_t)(bb * H_HEADS + hh) * L_SEQ + l;
        qkvh[(size_t)t * RROWS * DHEAD + row * DHEAD + dd] = (half_t)(acc[mi][j][r] + bias);
      }
  }
}

// ---------------- L2-normalize q (with learned scale) and k, in place ----------------
__global__ __launch_bounds__(256)
void l2norm_qk_kernel(half_t* __restrict__ qkvh, const float* __restrict__ scale_mul) {
  const int wid = (blockIdx.x * 256 + threadIdx.x) >> 5;   // one wave per 64-elem row
  const int lane = threadIdx.x & 31;
  if (wid >= 2 * RROWS) return;                             // q rows then k rows (contiguous)
  half_t* row = qkvh + (size_t)wid * DHEAD;
  float a = (float)row[lane * 2];
  float b = (float)row[lane * 2 + 1];
  float ss = a * a + b * b;
#pragma unroll
  for (int msk = 1; msk < 32; msk <<= 1) ss += __shfl_xor(ss, msk, 32);
  float scl = 1.0f / fmaxf(sqrtf(ss), 1e-12f);
  if (wid < RROWS) {                                        // q: apply exp(min(scale, log 100))
    const int hh = (wid / L_SEQ) & (H_HEADS - 1);
    scl *= __expf(fminf(scale_mul[hh], 4.6051702f));
  }
  row[lane * 2]     = (half_t)(a * scl);
  row[lane * 2 + 1] = (half_t)(b * scl);
}

// ---------------- Tensor Data Mover: 64x64 f16 tile -> LDS with 72-half padded stride ----------
// pad_interval=4 (32 DWORDs = one 64-half row), pad_amount=3 (4 DWORDs = 8 halves) => stride 72.
__device__ __forceinline__ void tdm_load_k_tile(unsigned lds_addr, const half_t* gptr) {
  unsigned long long ga = (unsigned long long)(size_t)gptr;
  u32x4 g0;
  g0[0] = 1u;                                   // count=1, user descriptor
  g0[1] = lds_addr;                             // LDS byte address
  g0[2] = (unsigned)ga;                         // global_addr[31:0]
  g0[3] = (unsigned)((ga >> 32) & 0x01FFFFFFu)  // global_addr[56:32]
          | 0x80000000u;                        // type=2 (bits 127:126)
  i32x8 g1;
  g1[0] = (1 << 16)      // data_size = 2B
        | (1 << 20)      // pad_enable
        | (4 << 22)      // pad_interval: 32 DWORDs
        | (3 << 25);     // pad_amount: 4 DWORDs
  g1[1] = (DHEAD & 0xFFFF) << 16;               // tensor_dim0[15:0] @ bits 63:48
  g1[2] = (int)((unsigned)L_SEQ << 16);         // tensor_dim0[31:16]=0, tensor_dim1[15:0]
  g1[3] = (64 << 16);                           // tensor_dim1[31:16]=0, tile_dim0=64
  g1[4] = 64;                                   // tile_dim1=64, tile_dim2=0
  g1[5] = DHEAD;                                // tensor_dim0_stride[31:0]
  g1[6] = 0;                                    // stride0[47:32]=0, stride1[15:0]=0
  g1[7] = 0;
  i32x4 gz = {0, 0, 0, 0};
#if __has_include(<hip/amd_detail/amd_gfx1250_TDM.h>)
  i32x8 gz8 = {0, 0, 0, 0, 0, 0, 0, 0};
  __builtin_amdgcn_tensor_load_to_lds(g0, g1, gz, gz, gz8, 0);
#else
  __builtin_amdgcn_tensor_load_to_lds(g0, g1, gz, gz, 0);
#endif
}

// ---------------- flash attention: softmax(q k^T + bias) v ----------------
#define KBLK 64
#define LDK  72
#define LDP  72

__global__ __launch_bounds__(128)
void attn_kernel(const half_t* __restrict__ qh, const half_t* __restrict__ kh,
                 const half_t* __restrict__ vh, const float* __restrict__ attn_bias,
                 half_t* __restrict__ aoh) {
  __shared__ __align__(16) half_t sK[2][KBLK * LDK];    // [key][d], TDM-filled, stride 72
  __shared__ __align__(16) half_t sVt[2][DHEAD * LDK];  // [d][key] (transposed for B-frags)
  __shared__ __align__(16) half_t sP[4][16 * LDP];      // per-wave P staging

  const int bh = blockIdx.y;
  const int b = bh >> 4, h = bh & 15;
  const int wave = threadIdx.x >> 5;
  const int lane = threadIdx.x & 31;
  const int ln = lane & 15, hif = lane >> 4;
  const int q0 = blockIdx.x * 64 + wave * 16;

  const half_t* qbase = qh + ((size_t)bh * L_SEQ + q0) * DHEAD;
  v16h qa0 = load_a_frag(qbase, DHEAD);        // d = 0..31
  v16h qa1 = load_a_frag(qbase + 32, DHEAD);   // d = 32..63

  v8f zero = {};
  v8f acc[4] = {zero, zero, zero, zero};       // O: 16 x 64 fp32
  float rowmax[8], rowsum[8];
#pragma unroll
  for (int r = 0; r < 8; ++r) { rowmax[r] = -1e30f; rowsum[r] = 0.0f; }

  const half_t* kbh = kh + (size_t)bh * L_SEQ * DHEAD;
  const half_t* vbh = vh + (size_t)bh * L_SEQ * DHEAD;
  const unsigned sK_lds0 = (unsigned)(size_t)&sK[0][0];
  const unsigned sK_lds1 = (unsigned)(size_t)&sK[1][0];

  // V copy helper values
  const int vkey = threadIdx.x >> 1;
  const int vhoff = (threadIdx.x & 1) * 32;

  // ---- prologue: fill buffer 0 with tile 0 ----
  if (wave == 0) tdm_load_k_tile(sK_lds0, kbh);
  {
    const half_t* gv = vbh + (size_t)vkey * DHEAD + vhoff;
#pragma unroll
    for (int i = 0; i < 32; ++i) sVt[0][(vhoff + i) * LDK + vkey] = gv[i];
  }
  if (wave == 0) __builtin_amdgcn_s_wait_tensorcnt(0);
  __syncthreads();

  const int NT = L_SEQ / KBLK;    // 32 key tiles
  for (int it = 0; it < NT; ++it) {
    const int cur = it & 1;
    const int kb = it * KBLK;

    // ---- prefetch tile it+1 into the other buffer (overlaps the WMMAs below) ----
    if (it + 1 < NT) {
      if (wave == 0)
        tdm_load_k_tile(cur ? sK_lds0 : sK_lds1, kbh + (size_t)(kb + KBLK) * DHEAD);
      const half_t* gv = vbh + (size_t)(kb + KBLK + vkey) * DHEAD + vhoff;
      half_t* vt = &sVt[1 - cur][0];
#pragma unroll
      for (int i = 0; i < 32; ++i) vt[(vhoff + i) * LDK + vkey] = gv[i];
    }

    const half_t* sKc = &sK[cur][0];
    const half_t* sVc = &sVt[cur][0];

    // scores S = Q K^T + bias : four 16x16 tiles (keys kt*16..)
    v8f s[4];
#pragma unroll
    for (int kt = 0; kt < 4; ++kt) {
      v8f c = zero;
      c = wmma_f16(qa0, load_b_frag(sKc + kt * 16 * LDK + 0, LDK), c);
      c = wmma_f16(qa1, load_b_frag(sKc + kt * 16 * LDK + 32, LDK), c);
#pragma unroll
      for (int r = 0; r < 8; ++r) {
        const int qrow = q0 + r + 8 * hif;
        const int kcol = kb + kt * 16 + ln;
        c[r] += attn_bias[(size_t)qrow * L_SEQ + kcol];
      }
      s[kt] = c;
    }

    // online softmax: fp32 stats, replicated across each 16-lane half
#pragma unroll
    for (int r = 0; r < 8; ++r) {
      float mx = fmaxf(fmaxf(s[0][r], s[1][r]), fmaxf(s[2][r], s[3][r]));
#pragma unroll
      for (int msk = 1; msk < 16; msk <<= 1) mx = fmaxf(mx, __shfl_xor(mx, msk, 32));
      const float mnew = fmaxf(rowmax[r], mx);
      const float corr = __expf(rowmax[r] - mnew);
      rowmax[r] = mnew;
      float ps = 0.0f;
#pragma unroll
      for (int kt = 0; kt < 4; ++kt) {
        const float p = __expf(s[kt][r] - mnew);
        s[kt][r] = p;
        ps += p;
      }
#pragma unroll
      for (int msk = 1; msk < 16; msk <<= 1) ps += __shfl_xor(ps, msk, 32);
      rowsum[r] = rowsum[r] * corr + ps;
#pragma unroll
      for (int j = 0; j < 4; ++j) acc[j][r] *= corr;
    }

    // stage P (f16) to LDS: C-layout -> row-major [q][key]
    half_t* pw = &sP[wave][0];
#pragma unroll
    for (int kt = 0; kt < 4; ++kt)
#pragma unroll
      for (int r = 0; r < 8; ++r)
        pw[(r + 8 * hif) * LDP + kt * 16 + ln] = (half_t)s[kt][r];

    // O += P @ V  (contraction over 64 keys = two 32-chunks)
    v16h pa0 = load_a_frag(pw, LDP);
    v16h pa1 = load_a_frag(pw + 32, LDP);
#pragma unroll
    for (int j = 0; j < 4; ++j) {
      acc[j] = wmma_f16(pa0, load_b_frag(sVc + (j * 16) * LDK + 0, LDK), acc[j]);
      acc[j] = wmma_f16(pa1, load_b_frag(sVc + (j * 16) * LDK + 32, LDK), acc[j]);
    }

    // prefetch complete + everyone done with 'cur' before it is overwritten next round
    if (wave == 0) __builtin_amdgcn_s_wait_tensorcnt(0);
    __syncthreads();
  }

  // normalize and write attention output as f16 in (B, L, C) layout
#pragma unroll
  for (int j = 0; j < 4; ++j)
#pragma unroll
    for (int r = 0; r < 8; ++r) {
      const int qrow = q0 + r + 8 * hif;
      const float o = acc[j][r] / fmaxf(rowsum[r], 1e-20f);
      aoh[((size_t)(b * L_SEQ + qrow)) * CDIM + h * DHEAD + j * 16 + ln] = (half_t)o;
    }
}

// ---------------- output projection: out = attn_out @ W_proj^T + b_proj (fp32 out) ----------------
__global__ __launch_bounds__(32)
void proj_gemm_kernel(const half_t* __restrict__ ah, const half_t* __restrict__ wh,
                      const float* __restrict__ bproj, float* __restrict__ out) {
  const int m0 = blockIdx.x * 32;
  const int n0 = blockIdx.y * 64;
  const int lane = threadIdx.x & 31;
  const int ln = lane & 15, hif = lane >> 4;

  v8f zero = {};
  v8f acc[2][4];
#pragma unroll
  for (int mi = 0; mi < 2; ++mi)
#pragma unroll
    for (int j = 0; j < 4; ++j) acc[mi][j] = zero;

  gemm_core_32x64(ah + (size_t)m0 * CDIM, wh + (size_t)n0 * CDIM, acc);

#pragma unroll
  for (int j = 0; j < 4; ++j) {
    const int n = n0 + 16 * j + ln;
    const float bias = bproj[n];
#pragma unroll
    for (int mi = 0; mi < 2; ++mi)
#pragma unroll
      for (int r = 0; r < 8; ++r) {
        const int m = m0 + mi * 16 + r + 8 * hif;
        out[(size_t)m * CDIM + n] = acc[mi][j][r] + bias;
      }
  }
}

// ---------------- host-side launch ----------------
extern "C" void kernel_launch(void* const* d_in, const int* in_sizes, int n_in,
                              void* d_out, int out_size, void* d_ws, size_t ws_size,
                              hipStream_t stream) {
  (void)in_sizes; (void)n_in; (void)out_size; (void)ws_size;
  const float* x         = (const float*)d_in[0];
  const float* attn_bias = (const float*)d_in[1];
  const float* W_qkv     = (const float*)d_in[2];
  const float* q_bias    = (const float*)d_in[3];
  const float* v_bias    = (const float*)d_in[4];
  const float* scale_mul = (const float*)d_in[5];
  const float* W_proj    = (const float*)d_in[6];
  const float* b_proj    = (const float*)d_in[7];
  float* out = (float*)d_out;

  // workspace layout (f16): xh | wqkvh | wprojh | qkv (q,k,v) | attn_out  ~= 48 MB
  half_t* xh     = (half_t*)d_ws;
  half_t* wqkvh  = xh     + (size_t)M1 * CDIM;
  half_t* wprojh = wqkvh  + (size_t)3 * CDIM * CDIM;
  half_t* qkvh   = wprojh + (size_t)CDIM * CDIM;
  half_t* aoh    = qkvh   + (size_t)3 * RROWS * DHEAD;

  // 1) fp32 -> f16 conversions
  {
    int n1 = M1 * CDIM, n2 = 3 * CDIM * CDIM, n3 = CDIM * CDIM;
    cvt_f32_f16_kernel<<<(n1 + 255) / 256, 256, 0, stream>>>(x, xh, n1);
    cvt_f32_f16_kernel<<<(n2 + 255) / 256, 256, 0, stream>>>(W_qkv, wqkvh, n2);
    cvt_f32_f16_kernel<<<(n3 + 255) / 256, 256, 0, stream>>>(W_proj, wprojh, n3);
  }

  // 2) QKV GEMM
  {
    dim3 grid(M1 / 32, (3 * CDIM) / 64);
    qkv_gemm_kernel<<<grid, 32, 0, stream>>>(xh, wqkvh, q_bias, v_bias, qkvh);
  }

  // 3) L2-norm q (scaled) and k
  {
    int waves = 2 * RROWS;
    l2norm_qk_kernel<<<(waves * 32 + 255) / 256, 256, 0, stream>>>(qkvh, scale_mul);
  }

  // 4) flash attention
  {
    dim3 grid(L_SEQ / 64, BATCH * H_HEADS);
    attn_kernel<<<grid, 128, 0, stream>>>(qkvh,
                                          qkvh + (size_t)RROWS * DHEAD,
                                          qkvh + (size_t)2 * RROWS * DHEAD,
                                          attn_bias, aoh);
  }

  // 5) output projection
  {
    dim3 grid(M1 / 32, CDIM / 64);
    proj_gemm_kernel<<<grid, 32, 0, stream>>>(aoh, wprojh, b_proj, out);
  }
}
